// MultiHeadAttention_53541062312519
// MI455X (gfx1250) — compile-verified, matrix operand order checked by reading, not run on hardware
//
#include <hip/hip_runtime.h>

// Problem constants (from reference): B=2, S=2048, E=1024, H=16, D=64
#define B_ 2
#define S_ 2048
#define E_ 1024
#define H_ 16
#define D_ 64

typedef __attribute__((ext_vector_type(16))) _Float16 v16h;
typedef __attribute__((ext_vector_type(8)))  float    v8f;
typedef __attribute__((ext_vector_type(4)))  unsigned int u32x4;
typedef __attribute__((ext_vector_type(4)))  int       i32x4;
typedef __attribute__((ext_vector_type(8)))  int       i32x8;

// Tensor Data Mover availability (device pass only; 5-arg on ROCm7.2/clang-22,
// 6-arg on amdgpu-toolchain/clang-23).
#if defined(__HIP_DEVICE_COMPILE__) && __has_builtin(__builtin_amdgcn_tensor_load_to_lds) && __has_builtin(__builtin_amdgcn_s_wait_tensorcnt)
#define USE_TDM 1
#else
#define USE_TDM 0
#endif

// A/B fragment packing for v_wmma_f32_16x16x32_f16 (wave32):
// lane L: row/col index = L&15, kb = (L<16 ? 0 : 8)
// elements 0..7  <- mem[kb + 0 .. kb + 7]
// elements 8..15 <- mem[kb + 16 .. kb + 23]
__device__ __forceinline__ v16h frag_from_f16(const _Float16* base) {
  v16h f;
#pragma unroll
  for (int i = 0; i < 8; ++i) { f[i] = base[i]; f[i + 8] = base[16 + i]; }
  return f;
}

__device__ __forceinline__ v16h frag_from_f32(const float* base) {
  v16h f;
#pragma unroll
  for (int i = 0; i < 8; ++i) {
    f[i]     = (_Float16)base[i];
    f[i + 8] = (_Float16)base[16 + i];
  }
  return f;
}

// Reduce across the 16 lanes sharing one C-matrix row (xor masks 1..8 stay
// inside each 16-lane half of the wave32).
__device__ __forceinline__ float rowmax16(float v) {
#pragma unroll
  for (int m = 1; m < 16; m <<= 1) v = fmaxf(v, __shfl_xor(v, m, 32));
  return v;
}
__device__ __forceinline__ float rowsum16(float v) {
#pragma unroll
  for (int m = 1; m < 16; m <<= 1) v += __shfl_xor(v, m, 32);
  return v;
}

#if USE_TDM
// Issue a TDM load of a 64x64 f16 tile from a 2D tensor (row stride
// d0_stride elements) into LDS at lds_ptr. Uniform (SGPR) operands; TDM
// ignores EXEC and is tracked by TENSORcnt.
__device__ __forceinline__ void tdm_load_tile64x64_f16(
    void* lds_ptr, const _Float16* gptr,
    unsigned tensor_d0, unsigned tensor_d1, unsigned d0_stride)
{
  const unsigned long long ga = (unsigned long long)(uintptr_t)gptr;
  const unsigned lds_addr = (unsigned)(uintptr_t)lds_ptr;  // LDS aperture: low 32 bits

  u32x4 g0;
  g0[0] = 1u;                                   // count=1 (valid user descriptor)
  g0[1] = lds_addr;                             // D#.lds_addr (bytes)
  g0[2] = (unsigned)(ga & 0xffffffffu);         // global_addr[31:0]
  g0[3] = (unsigned)((ga >> 32) & 0x1ffffffu)   // global_addr[56:32]
        | (2u << 30);                           // type = 2 ("image")

  i32x8 g1;
  g1[0] = (int)(1u << 16);                      // wg_mask=0, data_size=1 (2 bytes)
  g1[1] = (int)(tensor_d0 << 16);               // tensor_dim0[15:0] @ bits63:48
  g1[2] = (int)((tensor_d0 >> 16) | (tensor_d1 << 16));   // dim0 hi | dim1 lo
  g1[3] = (int)((tensor_d1 >> 16) | (64u << 16));         // dim1 hi | tile_dim0=64
  g1[4] = (int)64u;                             // tile_dim1=64, tile_dim2=0
  g1[5] = (int)d0_stride;                       // tensor_dim0_stride[31:0]
  g1[6] = 0;                                    // stride hi | dim1_stride lo
  g1[7] = 0;

  i32x4 z4 = {};
#if defined(__clang_major__) && (__clang_major__ >= 23)
  i32x8 z8 = {};
  __builtin_amdgcn_tensor_load_to_lds(g0, g1, z4, z4, z8, 0);
#else
  __builtin_amdgcn_tensor_load_to_lds(g0, g1, z4, z4, 0);
#endif
}
#endif  // USE_TDM

// ---------------------------------------------------------------------------
// Kernel 1: per-head linear projections  out = (x @ W^T + b) [* 1/sqrt(D) for Q]
// One wave per 16-row tile; 8 WMMAs per tile. Q/K stored (bh, s, d); V stored
// transposed (bh, e, s) via branch-free stride-based addressing so the
// attention V tile is a plain strided 2D tile for the TDM.
// ---------------------------------------------------------------------------
__global__ __launch_bounds__(128) void qkv_proj_kernel(
    const float* __restrict__ q, const float* __restrict__ k, const float* __restrict__ v,
    const float* __restrict__ Wq, const float* __restrict__ bq,
    const float* __restrict__ Wk, const float* __restrict__ bk,
    const float* __restrict__ Wv, const float* __restrict__ bv,
    _Float16* __restrict__ Qp, _Float16* __restrict__ Kp, _Float16* __restrict__ VpT)
{
  const int which = blockIdx.y;          // 0=Q, 1=K, 2=V
  const float* X; const float* W; const float* bias; _Float16* dst; float scale;
  if (which == 0)      { X = q; W = Wq; bias = bq; dst = Qp;  scale = 0.125f; }
  else if (which == 1) { X = k; W = Wk; bias = bk; dst = Kp;  scale = 1.0f;  }
  else                 { X = v; W = Wv; bias = bv; dst = VpT; scale = 1.0f;  }

  // Branch-free output addressing: (bh, s, d) for Q/K, (bh, e, s) for V.
  const size_t rstride = (which == 2) ? (size_t)1  : (size_t)D_;  // step per s-row
  const size_t estride = (which == 2) ? (size_t)S_ : (size_t)1;   // step per e-col

  const int lane   = threadIdx.x & 31;
  const int wave   = threadIdx.x >> 5;
  const int tile   = blockIdx.x * 4 + wave;   // 0 .. B*H*S/16 - 1 (4096)
  const int head   = tile >> 7;               // b*H + h (128 s-tiles per head)
  const int s0     = (tile & 127) << 4;
  const int b      = head >> 4;
  const int h      = head & 15;

  const int n     = lane & 15;
  const int kb    = (lane < 16) ? 0 : 8;
  const int half8 = kb;

  // A = x rows (16 x 64 over K=d)
  const float* xrow = X + ((size_t)b * S_ + (size_t)(s0 + n)) * E_ + h * D_;
  v16h a0 = frag_from_f32(xrow + kb);
  v16h a1 = frag_from_f32(xrow + kb + 32);

  const size_t hbase = (size_t)head * S_ * D_;

#pragma unroll
  for (int nt = 0; nt < 4; ++nt) {
    const int e = nt * 16 + n;
    // B[k=d][n=e] = W[e, d]
    const float* wrow = W + (size_t)e * D_;
    v16h b0 = frag_from_f32(wrow + kb);
    v16h b1 = frag_from_f32(wrow + kb + 32);

    v8f c = {};
    c = __builtin_amdgcn_wmma_f32_16x16x32_f16(false, a0, false, b0, (short)0, c, false, false);
    c = __builtin_amdgcn_wmma_f32_16x16x32_f16(false, a1, false, b1, (short)0, c, false, false);

    const float bval = bias[e];
    // Per-nt base pointer; unrolled r offsets fold to immediates * rstride.
    _Float16* dptr = dst + hbase + (size_t)e * estride
                         + (size_t)(s0 + half8) * rstride;
#pragma unroll
    for (int r = 0; r < 8; ++r) {
      const float val = (c[r] + bval) * scale;
      dptr[(size_t)r * rstride] = (_Float16)val;
    }
  }
}

// ---------------------------------------------------------------------------
// Kernel 2: fused flash attention.
// Block = 4 waves, each wave owns a 16-row Q stripe of a 64-row Q tile.
// K/V tiles streamed into double-buffered LDS by the Tensor Data Mover
// (pipelined: tile i+1 DMA overlaps tile i compute), waited with
// s_wait_tensorcnt. Per tile: QK^T (8 WMMA) -> mask -> online softmax ->
// P staged through LDS -> PV (8 WMMA).
// ---------------------------------------------------------------------------
__global__ __launch_bounds__(128) void flash_attn_kernel(
    const _Float16* __restrict__ Qp, const _Float16* __restrict__ Kp,
    const _Float16* __restrict__ VpT, const int* __restrict__ mask,
    float* __restrict__ out)
{
  __shared__ __align__(16) _Float16 Kt[2][64 * 64];  // K tile, row-major [key][d]
  __shared__ __align__(16) _Float16 Vt[2][64 * 64];  // V tile, [e][key]
  __shared__ __align__(16) _Float16 Pt[4][16 * 64];  // per-wave P staging

  const int bh   = blockIdx.x;        // b*H + h
  const int q0   = blockIdx.y * 64;
  const int tid  = threadIdx.x;
  const int lane = tid & 31;
  const int wave = tid >> 5;
  const int n     = lane & 15;
  const int half8 = (lane < 16) ? 0 : 8;
  const int kb    = half8;

  const size_t headBase = (size_t)bh * S_ * D_;   // same extent for Kp and VpT

  // Q A-fragments, resident across the KV loop (1/sqrt(D) folded in)
  const _Float16* qrow = Qp + headBase + (size_t)(q0 + wave * 16 + n) * D_;
  v16h qa0 = frag_from_f16(qrow + kb);
  v16h qa1 = frag_from_f16(qrow + kb + 32);

  float mrun[8], lrun[8];
  v8f o[4];
  const v8f vzero = {};
#pragma unroll
  for (int r = 0; r < 8; ++r) { mrun[r] = -3.0e38f; lrun[r] = 0.0f; }
#pragma unroll
  for (int t = 0; t < 4; ++t) o[t] = vzero;

  constexpr int NT = S_ / 64;  // 32 kv tiles

#if USE_TDM
  if (wave == 0) {  // prime the pipeline: tile 0 -> buffer 0
    tdm_load_tile64x64_f16(&Kt[0][0], Kp  + headBase, 64u,  (unsigned)S_, 64u);
    tdm_load_tile64x64_f16(&Vt[0][0], VpT + headBase, (unsigned)S_, 64u, (unsigned)S_);
  }
#endif

  for (int it = 0; it < NT; ++it) {
    const int p  = it & 1;
    const int kv = it * 64;

    __syncthreads();  // all waves done with both buffers' previous contents

#if USE_TDM
    if (wave == 0) {
      if (it + 1 < NT) {
        const int kv1 = kv + 64;
        tdm_load_tile64x64_f16(&Kt[1 - p][0], Kp  + headBase + (size_t)kv1 * D_,
                               64u, (unsigned)S_, 64u);
        tdm_load_tile64x64_f16(&Vt[1 - p][0], VpT + headBase + (size_t)kv1,
                               (unsigned)S_, 64u, (unsigned)S_);
        __builtin_amdgcn_s_wait_tensorcnt(2);  // tile `it` (oldest 2 ops) done
      } else {
        __builtin_amdgcn_s_wait_tensorcnt(0);
      }
    }
#else
    {
      // Fallback: cooperative vector copies (both tiles are strided 2D, rows
      // of 128B). K tile is fully contiguous; V tile rows have stride S.
      const uint4* ks4 = (const uint4*)(Kp + headBase + (size_t)kv * D_);
      uint4* kd4 = (uint4*)&Kt[p][0];
#pragma unroll
      for (int i = 0; i < 4; ++i) kd4[tid + i * 128] = ks4[tid + i * 128];

      const int e  = tid >> 1;
      const int c0 = (tid & 1) * 32;
      const uint4* vs4 = (const uint4*)(VpT + headBase + (size_t)e * S_ + kv + c0);
      uint4* vd4 = (uint4*)&Vt[p][e * 64 + c0];
#pragma unroll
      for (int i = 0; i < 4; ++i) vd4[i] = vs4[i];
    }
#endif
    __syncthreads();  // tile `it` visible to all waves

    // --- scores = Q . K^T (16 x 64 per wave) ---
    v8f sc[4];
#pragma unroll
    for (int nt = 0; nt < 4; ++nt) {
      const _Float16* kbase = &Kt[p][(nt * 16 + n) * 64 + kb];
      v16h b0 = frag_from_f16(kbase);
      v16h b1 = frag_from_f16(kbase + 32);
      v8f c = {};
      c = __builtin_amdgcn_wmma_f32_16x16x32_f16(false, qa0, false, b0, (short)0, c, false, false);
      c = __builtin_amdgcn_wmma_f32_16x16x32_f16(false, qa1, false, b1, (short)0, c, false, false);
      sc[nt] = c;
    }

    // --- mask (reference: masked scores = -1e20 exactly) ---
#pragma unroll
    for (int r = 0; r < 8; ++r) {
      const int qs = q0 + wave * 16 + r + half8;
      const int* mq = mask + (size_t)qs * S_ + kv;
#pragma unroll
      for (int nt = 0; nt < 4; ++nt) {
        const int mv = mq[nt * 16 + n];
        sc[nt][r] = (mv == 0) ? -1.0e20f : sc[nt][r];
      }
    }

    // --- online softmax ---
    float corr[8];
#pragma unroll
    for (int r = 0; r < 8; ++r) {
      float vmax = fmaxf(fmaxf(sc[0][r], sc[1][r]), fmaxf(sc[2][r], sc[3][r]));
      vmax = rowmax16(vmax);
      const float mnew = fmaxf(mrun[r], vmax);
      corr[r] = __expf(mrun[r] - mnew);
      mrun[r] = mnew;
      float psum = 0.0f;
#pragma unroll
      for (int nt = 0; nt < 4; ++nt) {
        const float pv = __expf(sc[nt][r] - mnew);
        sc[nt][r] = pv;
        psum += pv;
      }
      lrun[r] = lrun[r] * corr[r] + rowsum16(psum);
    }

    // rescale accumulator, stage P as f16 in LDS (C-layout -> A-layout)
#pragma unroll
    for (int t = 0; t < 4; ++t)
#pragma unroll
      for (int r = 0; r < 8; ++r) o[t][r] *= corr[r];

#pragma unroll
    for (int nt = 0; nt < 4; ++nt)
#pragma unroll
      for (int r = 0; r < 8; ++r)
        Pt[wave][(r + half8) * 64 + nt * 16 + n] = (_Float16)sc[nt][r];

    __syncthreads();  // (Pt is per-wave and same-wave LDS is in-order; barrier
                      //  also keeps tile-buffer lifetimes simple)

    // --- O += P . V ---
#pragma unroll
    for (int ks = 0; ks < 2; ++ks) {
      v16h pa = frag_from_f16(&Pt[wave][n * 64 + kb + ks * 32]);
#pragma unroll
      for (int et = 0; et < 4; ++et) {
        v16h vb = frag_from_f16(&Vt[p][(et * 16 + n) * 64 + kb + ks * 32]);
        o[et] = __builtin_amdgcn_wmma_f32_16x16x32_f16(false, pa, false, vb, (short)0, o[et], false, false);
      }
    }
  }

  // --- epilogue: O / l, scatter back to (b, s, h*64 + e) fp32 ---
  const int b = bh >> 4;
  const int h = bh & 15;
#pragma unroll
  for (int r = 0; r < 8; ++r) {
    const int qs = q0 + wave * 16 + r + half8;
    const float inv = 1.0f / lrun[r];
    float* orow = out + ((size_t)b * S_ + qs) * E_ + h * D_;
#pragma unroll
    for (int et = 0; et < 4; ++et)
      orow[et * 16 + n] = o[et][r] * inv;
  }
}

// ---------------------------------------------------------------------------
// Launch: projection (fp32 -> f16 Q/K row-major + V head-transposed in d_ws,
// ~24 MB), then fused flash attention. Both on `stream`; graph-capture safe.
// ---------------------------------------------------------------------------
extern "C" void kernel_launch(void* const* d_in, const int* in_sizes, int n_in,
                              void* d_out, int out_size, void* d_ws, size_t ws_size,
                              hipStream_t stream) {
  (void)in_sizes; (void)n_in; (void)out_size; (void)ws_size;

  const float* q    = (const float*)d_in[0];
  const float* k    = (const float*)d_in[1];
  const float* v    = (const float*)d_in[2];
  const int*   mask = (const int*)d_in[3];
  const float* Wq   = (const float*)d_in[4];
  const float* bq   = (const float*)d_in[5];
  const float* Wk   = (const float*)d_in[6];
  const float* bk   = (const float*)d_in[7];
  const float* Wv   = (const float*)d_in[8];
  const float* bv   = (const float*)d_in[9];
  float* out = (float*)d_out;

  const size_t perTensor = (size_t)B_ * H_ * S_ * D_;  // 4,194,304 halves (8 MB)
  _Float16* Qp  = (_Float16*)d_ws;
  _Float16* Kp  = Qp + perTensor;
  _Float16* VpT = Kp + perTensor;

  // B*H*S/16 = 4096 wave-tiles, 4 waves per block, x3 tensors
  qkv_proj_kernel<<<dim3(1024, 3), 128, 0, stream>>>(
      q, k, v, Wq, bq, Wk, bk, Wv, bv, Qp, Kp, VpT);

  // (B*H, S/64) blocks of 4 waves
  flash_attn_kernel<<<dim3(B_ * H_, S_ / 64), 128, 0, stream>>>(
      Qp, Kp, VpT, mask, out);
}
